// Competition_4260607558106
// MI455X (gfx1250) — 8-bit matrix operand order repaired and checked by reading, compiled, not run on hardware
//
#include <hip/hip_runtime.h>
#include <math.h>

#define B_ 2
#define H_ 256
#define W_ 256
#define N_ 65536
#define Q_ 32
#define P_ 128

typedef __attribute__((ext_vector_type(2))) float v2f;
typedef __attribute__((ext_vector_type(8))) float v8f;
typedef __attribute__((ext_vector_type(8))) int   v8i;

#define ROWS_PER_BLOCK 1024
#define NTILES (ROWS_PER_BLOCK / 16)

#define LDS_OFF(p) ((unsigned)(uintptr_t)(p))

__device__ __forceinline__ void async_tile_load(const float* __restrict__ gbase,
                                                int n0, unsigned lds_off, int tid) {
  if (tid < 128) {
    const int r   = tid >> 3;
    const int seg = tid & 7;
    const unsigned gvoff = (unsigned)((n0 + r) * (Q_ * 4) + seg * 16);
    const unsigned loff  = lds_off + (unsigned)(r * (Q_ * 4) + seg * 16);
    asm volatile("global_load_async_to_lds_b128 %0, %1, %2"
                 :: "v"(loff), "v"(gvoff), "s"(gbase)
                 : "memory");
  }
}

__device__ __forceinline__ void wait_async_le1() {
  asm volatile("s_wait_asynccnt 0x1" ::: "memory");
}
__device__ __forceinline__ void wait_async_0() {
  asm volatile("s_wait_asynccnt 0x0" ::: "memory");
}

__global__ __launch_bounds__(256)
void masks_intersect_kernel(const float* __restrict__ plateau,
                            const float* __restrict__ phenotypes,
                            int* __restrict__ Iacc) {
  __shared__ float         s_k[P_ * Q_];
  __shared__ float         s_row[2][16 * Q_];
  __shared__ float         s_inv[16];
  __shared__ unsigned char s_mbT[P_ * 64];

  const int b    = blockIdx.y;
  const int tid  = threadIdx.x;
  const int wave = tid >> 5;
  const int lane = tid & 31;
  const int g    = lane >> 4;
  const int ln   = lane & 15;

  const float* gb = plateau + (size_t)b * N_ * Q_;
  const int n_base = blockIdx.x * ROWS_PER_BLOCK;
  const unsigned rb_off[2] = {LDS_OFF(&s_row[0][0]), LDS_OFF(&s_row[1][0])};

  async_tile_load(gb, n_base, rb_off[0], tid);

  if (tid < P_) {
    const float* src = phenotypes + ((size_t)b * P_ + tid) * Q_;
    float ss = 0.f;
    #pragma unroll
    for (int q = 0; q < Q_; ++q) { float v = src[q]; ss += v * v; }
    const float inv = 1.0f / fmaxf(sqrtf(ss), 1e-6f);
    #pragma unroll
    for (int q = 0; q < Q_; ++q) s_k[tid * Q_ + q] = src[q] * inv;
  }

  v8i acc[8];
  #pragma unroll
  for (int t = 0; t < 8; ++t) {
    #pragma unroll
    for (int j = 0; j < 8; ++j) acc[t][j] = 0;
  }
  __syncthreads();

  for (int c = 0; c < 16; ++c) {
    for (int sub = 0; sub < 4; ++sub) {
      const int t = c * 4 + sub;
      if (t + 1 < NTILES) {
        async_tile_load(gb, n_base + (t + 1) * 16, rb_off[(t + 1) & 1], tid);
        wait_async_le1();
      } else {
        wait_async_0();
      }
      __syncthreads();
      const float* rb = &s_row[t & 1][0];

      if (tid < 16) {
        float ss = 0.f;
        #pragma unroll
        for (int q = 0; q < Q_; ++q) { float v = rb[tid * Q_ + q]; ss += v * v; }
        s_inv[tid] = 1.0f / fmaxf(sqrtf(ss), 1e-6f);
      }
      __syncthreads();

      v8f cm = {};
      const float ainv = s_inv[ln];
      #pragma unroll
      for (int s = 0; s < 8; ++s) {
        const int k = 4 * s + 2 * g;
        v2f a, bb;
        a.x  = rb[ln * Q_ + k]     * ainv;
        a.y  = rb[ln * Q_ + k + 1] * ainv;
        bb.x = s_k[(wave * 16 + ln) * Q_ + k];
        bb.y = s_k[(wave * 16 + ln) * Q_ + k + 1];
        cm = __builtin_amdgcn_wmma_f32_16x16x4_f32(false, a, false, bb,
                                                   (short)0, cm, false, false);
      }
      unsigned int lo = 0, hi = 0;
      #pragma unroll
      for (int j = 0; j < 4; ++j) lo |= (cm[j]     > 0.5f ? 1u : 0u) << (8 * j);
      #pragma unroll
      for (int j = 0; j < 4; ++j) hi |= (cm[j + 4] > 0.5f ? 1u : 0u) << (8 * j);
      const int p = wave * 16 + ln;
      *(unsigned int*)&s_mbT[p * 64 + sub * 16 + 8 * g]     = lo;
      *(unsigned int*)&s_mbT[p * 64 + sub * 16 + 8 * g + 4] = hi;
      __syncthreads();
    }

    v8i af;
    #pragma unroll
    for (int j = 0; j < 8; ++j) {
      const int k0 = (j >> 1) * 16 + (j & 1) * 4 + 8 * g;
      af[j] = *(const int*)&s_mbT[(wave * 16 + ln) * 64 + k0];
    }
    #pragma unroll
    for (int pj = 0; pj < 8; ++pj) {
      v8i bf;
      #pragma unroll
      for (int j = 0; j < 8; ++j) {
        const int k = (j >> 1) * 16 + (j & 1) * 4 + 8 * g;
        bf[j] = *(const int*)&s_mbT[(pj * 16 + ln) * 64 + k];
      }
      acc[pj] = __builtin_amdgcn_wmma_i32_16x16x64_iu8(false, af, false, bf,
                                                       acc[pj], false, false);
    }
    __syncthreads();
  }

  int* Ib = Iacc + (size_t)b * P_ * P_;
  #pragma unroll
  for (int pj = 0; pj < 8; ++pj) {
    #pragma unroll
    for (int j = 0; j < 8; ++j) {
      const int row = wave * 16 + j + 8 * g;
      const int col = pj * 16 + ln;
      atomicAdd(&Ib[row * P_ + col], acc[pj][j]);
    }
  }
}

__global__ __launch_bounds__(256)
void compete_kernel(const float* __restrict__ plateau,
                    const float* __restrict__ phenotypes,
                    const float* __restrict__ positions,
                    const float* __restrict__ alive,
                    const int* __restrict__ Iacc,
                    float* __restrict__ alive_new) {
  __shared__ float s_fit[B_ * P_];
  __shared__ float s_area[B_ * P_];
  __shared__ int   s_win[B_ * P_];
  const int tid = threadIdx.x;
  const int b = tid >> 7;
  const int p = tid & (P_ - 1);

  float kn[Q_];
  {
    const float* ph = phenotypes + ((size_t)b * P_ + p) * Q_;
    float ss = 0.f;
    #pragma unroll
    for (int q = 0; q < Q_; ++q) { kn[q] = ph[q]; ss += kn[q] * kn[q]; }
    const float inv = 1.0f / fmaxf(sqrtf(ss), 1e-6f);
    #pragma unroll
    for (int q = 0; q < Q_; ++q) kn[q] *= inv;
  }

  const float* pos = positions + ((size_t)b * P_ + p) * 2;
  float h = (pos[0] + 1.0f) * (H_ * 0.5f);
  float w = (pos[1] + 1.0f) * (W_ * 0.5f);
  h = fminf(fmaxf(h, 0.0f), (float)(H_ - 1));
  w = fminf(fmaxf(w, 0.0f), (float)(W_ - 1));
  const float hf = floorf(h), wf = floorf(w), hc = ceilf(h), wc = ceilf(w);
  const float brw = (h - hf) * (w - wf);
  const float blw = (h - hf) * (wc - w);
  const float trw = (hc - h) * (w - wf);
  const float tlw = (hc - h) * (wc - w);
  const float* base = plateau + (size_t)b * N_ * Q_;
  const float* r00 = base + (((int)hf) * W_ + (int)wf) * Q_;
  const float* r01 = base + (((int)hf) * W_ + (int)wc) * Q_;
  const float* r10 = base + (((int)hc) * W_ + (int)wf) * Q_;
  const float* r11 = base + (((int)hc) * W_ + (int)wc) * Q_;
  float pv[Q_], pss = 0.f;
  #pragma unroll
  for (int q = 0; q < Q_; ++q) {
    pv[q] = tlw * r00[q] + trw * r01[q] + blw * r10[q] + brw * r11[q];
    pss += pv[q] * pv[q];
  }
  const float pinv = 1.0f / fmaxf(sqrtf(pss), 1e-6f);
  float fit = 0.f;
  #pragma unroll
  for (int q = 0; q < Q_; ++q) fit += kn[q] * pv[q] * pinv;

  s_fit[tid]  = fit;
  s_win[tid]  = (alive[(size_t)b * P_ + p] > 0.5f) ? 1 : 0;
  s_area[tid] = (float)Iacc[(size_t)b * P_ * P_ + p * P_ + p];
  __syncthreads();

  const int* Ib = Iacc + (size_t)b * P_ * P_;
  const float sp = s_area[tid];
  const float fp = fit;
  const int winp = s_win[tid];
  bool killed = false;
  for (int q = 0; q < P_; ++q) {
    if (q == p) continue;
    const float Ipq = (float)Ib[p * P_ + q];
    const float U   = sp + s_area[b * P_ + q] - Ipq;
    const float iou = Ipq / fmaxf(U, 1e-6f);
    if (!(iou > 0.2f)) continue;
    const float fq = s_fit[b * P_ + q];
    const int winq = s_win[b * P_ + q];
    const bool k1 = (fp < fq) && !(winp && !winq);
    const bool k2 = (!winp) && winq;
    if (k1 || k2) { killed = true; break; }
  }
  alive_new[(size_t)b * P_ + p] = killed ? 0.0f : 1.0f;
}

__global__ __launch_bounds__(256)
void output_kernel(const float* __restrict__ plateau,
                   const float* __restrict__ phenotypes,
                   const float* __restrict__ alive_new,
                   float* __restrict__ out) {
  __shared__ float s_k[P_ * Q_];
  __shared__ float s_row[2][16 * Q_];
  __shared__ float s_inv[16];
  __shared__ float s_alive[P_];

  const int b    = blockIdx.y;
  const int tid  = threadIdx.x;
  const int wave = tid >> 5;
  const int lane = tid & 31;
  const int g    = lane >> 4;
  const int ln   = lane & 15;

  const float* gb = plateau + (size_t)b * N_ * Q_;
  const int n_base = blockIdx.x * ROWS_PER_BLOCK;
  const unsigned rb_off[2] = {LDS_OFF(&s_row[0][0]), LDS_OFF(&s_row[1][0])};

  async_tile_load(gb, n_base, rb_off[0], tid);

  if (tid < P_) {
    const float* src = phenotypes + ((size_t)b * P_ + tid) * Q_;
    float ss = 0.f;
    #pragma unroll
    for (int q = 0; q < Q_; ++q) { float v = src[q]; ss += v * v; }
    const float inv = 1.0f / fmaxf(sqrtf(ss), 1e-6f);
    #pragma unroll
    for (int q = 0; q < Q_; ++q) s_k[tid * Q_ + q] = src[q] * inv;
    s_alive[tid] = alive_new[(size_t)b * P_ + tid];
  }
  __syncthreads();

  const float ap = s_alive[wave * 16 + ln];
  for (int t = 0; t < NTILES; ++t) {
    const int n0 = n_base + t * 16;
    if (t + 1 < NTILES) {
      async_tile_load(gb, n_base + (t + 1) * 16, rb_off[(t + 1) & 1], tid);
      wait_async_le1();
    } else {
      wait_async_0();
    }
    __syncthreads();
    const float* rb = &s_row[t & 1][0];

    if (tid < 16) {
      float ss = 0.f;
      #pragma unroll
      for (int q = 0; q < Q_; ++q) { float v = rb[tid * Q_ + q]; ss += v * v; }
      s_inv[tid] = 1.0f / fmaxf(sqrtf(ss), 1e-6f);
    }
    __syncthreads();

    v8f cm = {};
    const float ainv = s_inv[ln];
    #pragma unroll
    for (int s = 0; s < 8; ++s) {
      const int k = 4 * s + 2 * g;
      v2f a, bb;
      a.x  = rb[ln * Q_ + k]     * ainv;
      a.y  = rb[ln * Q_ + k + 1] * ainv;
      bb.x = s_k[(wave * 16 + ln) * Q_ + k];
      bb.y = s_k[(wave * 16 + ln) * Q_ + k + 1];
      cm = __builtin_amdgcn_wmma_f32_16x16x4_f32(false, a, false, bb,
                                                 (short)0, cm, false, false);
    }
    #pragma unroll
    for (int j = 0; j < 8; ++j) {
      const int row = n0 + j + 8 * g;
      out[((size_t)b * N_ + row) * P_ + wave * 16 + ln] = fmaxf(cm[j], 0.0f) * ap;
    }
    __syncthreads();
  }
}

extern "C" void kernel_launch(void* const* d_in, const int* in_sizes, int n_in,
                              void* d_out, int out_size, void* d_ws, size_t ws_size,
                              hipStream_t stream) {
  const float* plateau   = (const float*)d_in[0];
  const float* phen      = (const float*)d_in[1];
  const float* positions = (const float*)d_in[2];
  const float* alive     = (const float*)d_in[3];
  float* out = (float*)d_out;

  int*   Iacc      = (int*)d_ws;
  float* alive_new = (float*)((char*)d_ws + (size_t)B_ * P_ * P_ * sizeof(int));

  hipMemsetAsync(d_ws, 0, (size_t)B_ * P_ * P_ * sizeof(int), stream);

  dim3 grid(N_ / ROWS_PER_BLOCK, B_);
  masks_intersect_kernel<<<grid, 256, 0, stream>>>(plateau, phen, Iacc);
  compete_kernel<<<dim3(1), 256, 0, stream>>>(plateau, phen, positions, alive,
                                              Iacc, alive_new);
  output_kernel<<<grid, 256, 0, stream>>>(plateau, phen, alive_new, out);
}
